// OptimizedLSTM_69191923139126
// MI455X (gfx1250) — matrix operand order repaired and checked by reading, not compiled, not run on hardware
//
#include <hip/hip_runtime.h>
#include <hip/hip_bf16.h>
#include <math.h>

typedef __attribute__((ext_vector_type(16))) __bf16 v16bf;
typedef __attribute__((ext_vector_type(8)))  float  v8f;

#define SEQ    512
#define BATCH  64
#define ISZ    512
#define HSZ    512
#define GSZ    2048          // 4*HSZ
#define NT_TOT 128           // GSZ/16
#define KT     16            // 512/32

// ---------------------------------------------------------------------------
// Fragment layout helpers (per CDNA5 ISA 7.12.2, wave32):
//  A (16x32 bf16, MxK): lane = (kk&8?16:0) + m ; elem e: k = ((e>>3)&1)*16 + (kk>>4? ...) 
//    encoded: for k-in-tile kk: hi=(kk>>3)&1, e=((kk>>4)&1)*8+(kk&7), lane=hi*16+m
//  B (32x16 bf16, KxN): lane = ((kk>>3)&1)*16 + n ; e = ((kk>>4)&1)*8+(kk&7) (same k order as A)
//  C/D (16x16 f32): vgpr v, lanes0-15: M=v,N=lane ; lanes16-31: M=v+8,N=lane-16
// Fragments are stored flat as [tile][lane][16 elems] so each lane loads its
// v16bf with contiguous 32-byte loads.
// ---------------------------------------------------------------------------

// ----------------------------- prep kernel --------------------------------
// Swizzle f32 weights -> bf16 fragment buffers; scatter h0 into h-fragment
// buffer 0; copy c0 into the f32 cell-state buffer.
__global__ __launch_bounds__(256) void lstm_prep(
    const float* __restrict__ w_ih, const float* __restrict__ w_hh,
    const float* __restrict__ h0,   const float* __restrict__ c0,
    __bf16* __restrict__ bih, __bf16* __restrict__ bhh,
    __bf16* __restrict__ hfrag0, float* __restrict__ c_state)
{
  int i = blockIdx.x * 256 + threadIdx.x;        // 0 .. ISZ*GSZ-1
  if (i < ISZ * GSZ) {
    int k  = i >> 11;          // source row (K)
    int n  = i & 2047;         // source col (N)
    int kt = k >> 5, kk = k & 31;
    int nt = n >> 4, nl = n & 15;
    int lane = nl + (((kk >> 3) & 1) << 4);
    int e    = ((kk >> 4) & 1) * 8 + (kk & 7);
    int pos  = ((kt * NT_TOT + nt) * 32 + lane) * 16 + e;
    bih[pos] = (__bf16)w_ih[i];
    bhh[pos] = (__bf16)w_hh[i];
  }
  if (i < BATCH * HSZ) {
    int b = i >> 9, j = i & 511;
    int kt = j >> 5, kk = j & 31;
    int hi = (kk >> 3) & 1;
    int e  = ((kk >> 4) & 1) * 8 + (kk & 7);
    int lane = hi * 16 + (b & 15);
    int mt   = b >> 4;
    hfrag0[((kt * 4 + mt) * 32 + lane) * 16 + e] = (__bf16)h0[i];
    c_state[i] = c0[i];
  }
}

// --------------------------- phase 1: x_proj -------------------------------
// One block per timestep s. Stage x[:,s,:] (64x512) into LDS in A-fragment
// order (bf16), then 8 waves do (4 mtiles)x(128 ntiles) 16x16 output tiles
// with v_wmma_f32_16x16x32_bf16. x_proj layout: [(s*64+b)*2048 + n].
__global__ __launch_bounds__(256) void lstm_xproj(
    const float* __restrict__ x, const __bf16* __restrict__ bih,
    float* __restrict__ xproj)
{
  __shared__ __bf16 ldsA[64 * ISZ];              // 64KB, fragment order
  const int s = blockIdx.x;

  for (int i = threadIdx.x; i < 64 * ISZ; i += 256) {
    int m = i >> 9, k = i & 511;                 // m = batch index
    float v = x[((size_t)m * SEQ + s) * ISZ + k];
    int mt = m >> 4, ml = m & 15;
    int kt = k >> 5, kk = k & 31;
    int hi = (kk >> 3) & 1;
    int e  = ((kk >> 4) & 1) * 8 + (kk & 7);
    int lane = hi * 16 + ml;
    ldsA[((kt * 4 + mt) * 32 + lane) * 16 + e] = (__bf16)v;
  }
  __syncthreads();

  const int lane   = threadIdx.x & 31;
  const int w      = threadIdx.x >> 5;
  const int mt     = w & 3;
  const int ntBase = (w >> 2) * 64;              // waves 0-3: nt 0-63, 4-7: 64-127
  const int mrow   = mt * 16 + (lane >> 4) * 8;
  const int ncol   = lane & 15;

  for (int grp = 0; grp < 8; ++grp) {
    v8f acc[8] = {};
    for (int kt = 0; kt < KT; ++kt) {
      v16bf a = *(const v16bf*)&ldsA[((kt * 4 + mt) * 32 + lane) * 16];
      const __bf16* bp =
          bih + ((size_t)(kt * NT_TOT + ntBase + grp * 8) * 32 + lane) * 16;
#pragma unroll
      for (int u = 0; u < 8; ++u) {
        v16bf bf = *(const v16bf*)(bp + (size_t)u * 512);
        acc[u] = __builtin_amdgcn_wmma_f32_16x16x32_bf16(
            false, a, false, bf, (short)0, acc[u], false, false);
      }
    }
#pragma unroll
    for (int u = 0; u < 8; ++u) {
      int n = (ntBase + grp * 8 + u) * 16 + ncol;
      float* op = xproj + (size_t)(s * 64) * GSZ + n;
#pragma unroll
      for (int v = 0; v < 8; ++v)
        op[(size_t)(mrow + v) * GSZ] = acc[u][v];
    }
  }
}

__device__ __forceinline__ float sigm(float v) {
  return 1.0f / (1.0f + __expf(-v));
}

// --------------------------- phase 2: recurrence ---------------------------
// Persistent kernel, 32 blocks x 256 threads. Block `blk` owns hidden columns
// [blk*16, blk*16+16) -> gate ntiles {g*32+blk : g=0..3}. Per step:
//   16 WMMA tiles (8 waves x 2) -> LDS -> gate activations -> c/h update ->
//   scatter new h (bf16, fragment order, double buffered) -> grid barrier.
__global__ __launch_bounds__(256) void lstm_recur(
    const float* __restrict__ xproj, const __bf16* __restrict__ bhh,
    const float* __restrict__ bias,
    __bf16* __restrict__ hfragA, __bf16* __restrict__ hfragB,
    float* __restrict__ c_state,
    float* __restrict__ out_seq, float* __restrict__ out_hT,
    float* __restrict__ out_cT, unsigned int* __restrict__ counter)
{
  __shared__ float ldsG[4 * 64 * 16];            // gates [g][b][j], 16KB
  const int tid  = threadIdx.x;
  const int lane = tid & 31;
  const int w    = tid >> 5;
  const int mt   = w & 3;                        // batch tile 0..3
  const int g0   = (w >> 2) * 2;                 // waves 0-3: gates 0,1 ; 4-7: 2,3
  const int blk  = blockIdx.x;                   // 0..31
  const int j0   = blk * 16;
  const unsigned int nb = gridDim.x;
  const int mrow = mt * 16 + (lane >> 4) * 8;
  const int ncol = lane & 15;

  for (int s = 0; s < SEQ; ++s) {
    const __bf16* hcur = (s & 1) ? hfragB : hfragA;
    __bf16*       hnxt = (s & 1) ? hfragA : hfragB;

#pragma unroll
    for (int gg = 0; gg < 2; ++gg) {
      const int g  = g0 + gg;
      const int nt = g * 32 + blk;
      v8f acc = {};
      for (int kt = 0; kt < KT; ++kt) {
        v16bf a = *(const v16bf*)(hcur + ((kt * 4 + mt) * 32 + lane) * 16);
        v16bf b = *(const v16bf*)(bhh + ((size_t)(kt * NT_TOT + nt) * 32 + lane) * 16);
        acc = __builtin_amdgcn_wmma_f32_16x16x32_bf16(
            false, a, false, b, (short)0, acc, false, false);
      }
#pragma unroll
      for (int v = 0; v < 8; ++v)
        ldsG[(g * 64 + mrow + v) * 16 + ncol] = acc[v];
    }
    __syncthreads();

    // elementwise gate update for the 64x16 owned slice
    for (int p = tid; p < BATCH * 16; p += 256) {
      const int b = p >> 4;
      const int j = p & 15;
      const int n = j0 + j;
      const size_t xrow = (size_t)(s * BATCH + b) * GSZ;
      float gi = ldsG[(0 * 64 + b) * 16 + j] + xproj[xrow + 0 * HSZ + n] + bias[0 * HSZ + n];
      float gf = ldsG[(1 * 64 + b) * 16 + j] + xproj[xrow + 1 * HSZ + n] + bias[1 * HSZ + n];
      float gc = ldsG[(2 * 64 + b) * 16 + j] + xproj[xrow + 2 * HSZ + n] + bias[2 * HSZ + n];
      float go = ldsG[(3 * 64 + b) * 16 + j] + xproj[xrow + 3 * HSZ + n] + bias[3 * HSZ + n];
      float it = sigm(gi), ft = sigm(gf), gt = tanhf(gc), ot = sigm(go);
      float c  = ft * c_state[b * HSZ + n] + it * gt;
      float h  = ot * tanhf(c);
      c_state[b * HSZ + n] = c;
      out_seq[((size_t)b * SEQ + s) * HSZ + n] = h;
      // scatter h into next-step A-fragment buffer (bf16)
      int kt = n >> 5, kk = n & 31;
      int hi = (kk >> 3) & 1;
      int e  = ((kk >> 4) & 1) * 8 + (kk & 7);
      int lf = hi * 16 + (b & 15);
      int mb = b >> 4;
      hnxt[((kt * 4 + mb) * 32 + lf) * 16 + e] = (__bf16)h;
      if (s == SEQ - 1) {
        out_hT[b * HSZ + n] = h;
        out_cT[b * HSZ + n] = c;
      }
    }

    // --- device-wide barrier (all 32 blocks resident) ---
    __threadfence();                              // release my h writes
    __syncthreads();
    if (tid == 0) {
      atomicAdd(counter, 1u);
      const unsigned int target = nb * (unsigned int)(s + 1);
      while (__hip_atomic_load(counter, __ATOMIC_RELAXED,
                               __HIP_MEMORY_SCOPE_AGENT) < target) {
        __builtin_amdgcn_s_sleep(1);
      }
    }
    __syncthreads();
    __threadfence();                              // acquire: invalidate L0/WGP$
  }
}

// ------------------------------- launcher ----------------------------------
extern "C" void kernel_launch(void* const* d_in, const int* in_sizes, int n_in,
                              void* d_out, int out_size, void* d_ws, size_t ws_size,
                              hipStream_t stream) {
  const float* x    = (const float*)d_in[0];
  const float* wih  = (const float*)d_in[1];
  const float* whh  = (const float*)d_in[2];
  const float* bias = (const float*)d_in[3];
  const float* h0   = (const float*)d_in[4];
  const float* c0   = (const float*)d_in[5];

  char* ws = (char*)d_ws;
  size_t off = 0;
  auto carve = [&](size_t bytes) -> void* {
    void* p = ws + off;
    off = (off + bytes + 255) & ~(size_t)255;
    return p;
  };
  unsigned int* counter = (unsigned int*)carve(256);
  __bf16* hfragA  = (__bf16*)carve((size_t)BATCH * HSZ * 2);        // 64KB
  __bf16* hfragB  = (__bf16*)carve((size_t)BATCH * HSZ * 2);        // 64KB
  float*  c_state = (float*) carve((size_t)BATCH * HSZ * 4);        // 128KB
  __bf16* bih     = (__bf16*)carve((size_t)ISZ * GSZ * 2);          // 2MB
  __bf16* bhh     = (__bf16*)carve((size_t)HSZ * GSZ * 2);          // 2MB
  float*  xproj   = (float*) carve((size_t)SEQ * BATCH * GSZ * 4);  // 256MB

  float* out_seq = (float*)d_out;                        // (B, S, H)
  float* out_hT  = out_seq + (size_t)BATCH * SEQ * HSZ;  // (1, B, H)
  float* out_cT  = out_hT + (size_t)BATCH * HSZ;         // (1, B, H)

  hipMemsetAsync(counter, 0, 256, stream);

  lstm_prep<<<dim3((ISZ * GSZ + 255) / 256), dim3(256), 0, stream>>>(
      wih, whh, h0, c0, bih, bhh, hfragA, c_state);

  lstm_xproj<<<dim3(SEQ), dim3(256), 0, stream>>>(x, bih, xproj);

  lstm_recur<<<dim3(32), dim3(256), 0, stream>>>(
      xproj, bhh, bias, hfragA, hfragB, c_state,
      out_seq, out_hT, out_cT, counter);
}
